// MoeSparseMoeBlock_32392643346957
// MI455X (gfx1250) — matrix operand order repaired and checked by reading, not compiled, not run on hardware
//
#include <hip/hip_runtime.h>
#include <hip/hip_bf16.h>

typedef __attribute__((ext_vector_type(16))) __bf16 v16bf;
typedef __attribute__((ext_vector_type(8)))  float  v8f;
typedef unsigned short u16;
typedef unsigned int   u32;

#define H_DIM 1024
#define E_NUM 8
#define I_DIM 2816
#define T_NUM 8192
#define LP    40          // LDS row pitch (elements) — breaks ds bank conflicts

// ---- workspace layout (bytes). Total required ~94 MB. ----
#define WS_COUNTS   0u                                // 8 ints
#define WS_BASES    128u                              // 9 ints
#define WS_TOKSLOT  1024u                             // 2T ints
#define WS_TOKEXP   (WS_TOKSLOT + 4u*2u*T_NUM)        // 2T ints
#define WS_TOKW     (WS_TOKEXP  + 4u*2u*T_NUM)        // 2T floats
#define WS_ROWTOK   (WS_TOKW    + 4u*2u*T_NUM)        // 2T ints (row -> token)
#define WS_ROWW     (WS_ROWTOK  + 4u*2u*T_NUM)        // 2T floats (row -> combine weight)
#define WS_INTER    (1u<<20)                          // 2T x I bf16  (~88 MB)

union FragBF { v16bf v; uint4 q[2]; };

__device__ __forceinline__ u32 pack2bf(float lo, float hi) {   // -> v_cvt_pk_bf16_f32
  union { __bf16 h[2]; u32 u; } p;
  p.h[0] = (__bf16)lo;
  p.h[1] = (__bf16)hi;
  return p.u;
}
__device__ __forceinline__ u16 f2bf(float f) {
  union { __bf16 h; u16 u; } p;
  p.h = (__bf16)f;
  return p.u;
}
__device__ __forceinline__ v8f wmma_bf16(const FragBF& a, const FragBF& b, v8f c) {
  return __builtin_amdgcn_wmma_f32_16x16x32_bf16(false, a.v, false, b.v, (short)0, c,
                                                 false, false);
}
// raw LDS byte offset of a generic pointer into __shared__ (addrspace(3) value)
__device__ __forceinline__ u32 lds_addr_of(const void* p) {
  return (u32)(uintptr_t)(__attribute__((address_space(3))) const void*)p;
}

// ---------------------------------------------------------------- utilities
__global__ void init_counts_kernel(int* cnts) {
  if (threadIdx.x < E_NUM) cnts[threadIdx.x] = 0;
}

__global__ void zero_out_kernel(float4* out4) {
  out4[(size_t)blockIdx.x * blockDim.x + threadIdx.x] = make_float4(0.f, 0.f, 0.f, 0.f);
}

__global__ void prefix_kernel(const int* __restrict__ cnts, int* __restrict__ bases) {
  if (threadIdx.x == 0) {
    int s = 0;
    for (int e = 0; e < E_NUM; ++e) { bases[e] = s; s += cnts[e]; }
    bases[E_NUM] = s;
  }
}

__global__ void scatter_kernel(const int* __restrict__ bases,
                               const int* __restrict__ tokSlot,
                               const int* __restrict__ tokExp,
                               const float* __restrict__ tokW,
                               int* __restrict__ rowTok,
                               float* __restrict__ rowW) {
  const int t = blockIdx.x * blockDim.x + threadIdx.x;
  if (t >= T_NUM) return;
#pragma unroll
  for (int k = 0; k < 2; ++k) {
    const int i   = 2 * t + k;
    const int row = bases[tokExp[i]] + tokSlot[i];
    rowTok[row] = t;
    rowW[row]   = tokW[i];
  }
}

// ---------------------------------------------------------------- router
__global__ __launch_bounds__(256)
void router_kernel(const float* __restrict__ x, const float* __restrict__ gw,
                   float* __restrict__ logits, int* cnts,
                   int* tokSlot, int* tokExp, float* tokW)
{
  const int lane = threadIdx.x & 31;
  const int wave = threadIdx.x >> 5;
  const int t    = blockIdx.x * 8 + wave;

  float acc[8] = {0.f,0.f,0.f,0.f,0.f,0.f,0.f,0.f};
  const float* xr = x + (size_t)t * H_DIM;
  for (int h = lane; h < H_DIM; h += 32) {
    const float xv = xr[h];
    const float4* g4 = (const float4*)(gw + h * E_NUM);
    const float4 a = g4[0], b = g4[1];
    acc[0] += xv * a.x; acc[1] += xv * a.y; acc[2] += xv * a.z; acc[3] += xv * a.w;
    acc[4] += xv * b.x; acc[5] += xv * b.y; acc[6] += xv * b.z; acc[7] += xv * b.w;
  }
#pragma unroll
  for (int off = 16; off > 0; off >>= 1) {
#pragma unroll
    for (int e = 0; e < 8; ++e) acc[e] += __shfl_xor(acc[e], off, 32);
  }
  if (lane == 0) {
    *(float4*)(logits + (size_t)t * 8)     = make_float4(acc[0], acc[1], acc[2], acc[3]);
    *(float4*)(logits + (size_t)t * 8 + 4) = make_float4(acc[4], acc[5], acc[6], acc[7]);
    float m = acc[0];
#pragma unroll
    for (int e = 1; e < 8; ++e) m = fmaxf(m, acc[e]);
    float p[8];
#pragma unroll
    for (int e = 0; e < 8; ++e) p[e] = __expf(acc[e] - m);
    int e1 = 0; float p1 = p[0];
#pragma unroll
    for (int e = 1; e < 8; ++e) if (p[e] > p1) { p1 = p[e]; e1 = e; }
    int e2 = -1; float p2 = -1.f;
#pragma unroll
    for (int e = 0; e < 8; ++e) if (e != e1 && p[e] > p2) { p2 = p[e]; e2 = e; }
    const float inv = 1.f / (p1 + p2);
    const int s1 = atomicAdd(&cnts[e1], 1);
    const int s2 = atomicAdd(&cnts[e2], 1);
    tokSlot[2*t]   = s1; tokExp[2*t]   = e1; tokW[2*t]   = p1 * inv;
    tokSlot[2*t+1] = s2; tokExp[2*t+1] = e2; tokW[2*t+1] = p2 * inv;
  }
}

// ---------------------------------------------------------------- gate/up GEMM
// inter[row,:] = silu(x @ Wg[e]) * (x @ Wu[e]); block tile 128x128, K-step 32,
// double-buffered LDS, 8 waves in a 4x2 grid (each wave: 32 rows x 64 cols).
__global__ __launch_bounds__(256)
void moe_gateup_kernel(const float* __restrict__ x,
                       const float* __restrict__ Wg_all,
                       const float* __restrict__ Wu_all,
                       const int* __restrict__ cnts,
                       const int* __restrict__ bases,
                       const int* __restrict__ rowTok,
                       u16* __restrict__ inter)
{
  const int e     = blockIdx.z;
  const int cnt   = cnts[e];
  const int mbase = blockIdx.y * 128;
  if (mbase >= cnt) return;                      // uniform early-exit
  const int n0   = blockIdx.x * 128;
  const int base = bases[e];
  const float* Wg = Wg_all + (size_t)e * H_DIM * I_DIM;
  const float* Wu = Wu_all + (size_t)e * H_DIM * I_DIM;

  __shared__ u16 sA [2][128 * LP];
  __shared__ u16 sBg[2][128 * LP];               // transposed [n][k]
  __shared__ u16 sBu[2][128 * LP];
  __shared__ int sTok[128];

  const int tid  = threadIdx.x;
  const int lane = tid & 31;
  const int wave = tid >> 5;
  const int lrow = lane & 15;
  const int hi   = lane >> 4;
  const int mw   = wave >> 1;                    // 0..3
  const int nw   = wave & 1;                     // 0..1

  if (tid < 128) {
    const int grow = mbase + tid;
    sTok[tid] = (grow < cnt) ? rowTok[base + grow] : 0;
  }
  __syncthreads();

  // A staging: thread -> (row, 16-elem half); pointer hoisted out of K loop
  const int  arow   = tid >> 1;
  const int  ahalf  = tid & 1;
  const bool avalid = (mbase + arow) < cnt;
  const float* aptr = x + (size_t)sTok[arow] * H_DIM + ahalf * 16;

  // B staging: thread -> (column n, k-quad base); 4x b64 transposed stores/matrix
  const int bn  = tid & 127;
  const int bkq = (tid >> 7) * 4;                // 0 or 4
  const float* gptr = Wg + n0 + bn;
  const float* uptr = Wu + n0 + bn;

  float af[16], gf[16], uf[16];

  auto load_tiles = [&](int K0) {
    if (avalid) {
      const float4* s4 = (const float4*)(aptr + K0);
#pragma unroll
      for (int j = 0; j < 4; ++j) {
        const float4 v = s4[j];
        af[4*j+0] = v.x; af[4*j+1] = v.y; af[4*j+2] = v.z; af[4*j+3] = v.w;
      }
    } else {
#pragma unroll
      for (int j = 0; j < 16; ++j) af[j] = 0.f;
    }
#pragma unroll
    for (int r = 0; r < 4; ++r) {
#pragma unroll
      for (int q = 0; q < 4; ++q) {
        const size_t gi = (size_t)(K0 + bkq + 8 * r + q) * I_DIM;
        gf[4*r+q] = gptr[gi];
        uf[4*r+q] = uptr[gi];
      }
    }
  };

  auto store_tiles = [&](int buf) {
    uint4* ad = (uint4*)&sA[buf][arow * LP + ahalf * 16];
    ad[0] = make_uint4(pack2bf(af[0],af[1]),  pack2bf(af[2],af[3]),
                       pack2bf(af[4],af[5]),  pack2bf(af[6],af[7]));
    ad[1] = make_uint4(pack2bf(af[8],af[9]),  pack2bf(af[10],af[11]),
                       pack2bf(af[12],af[13]),pack2bf(af[14],af[15]));
#pragma unroll
    for (int r = 0; r < 4; ++r) {
      const int ks = bkq + 8 * r;
      *(uint2*)&sBg[buf][bn * LP + ks] =
          make_uint2(pack2bf(gf[4*r],gf[4*r+1]), pack2bf(gf[4*r+2],gf[4*r+3]));
      *(uint2*)&sBu[buf][bn * LP + ks] =
          make_uint2(pack2bf(uf[4*r],uf[4*r+1]), pack2bf(uf[4*r+2],uf[4*r+3]));
    }
  };

  v8f accg[2][4] = {};
  v8f accu[2][4] = {};

  load_tiles(0);
  for (int kk = 0; kk < H_DIM / 32; ++kk) {
    const int buf = kk & 1;
    store_tiles(buf);
    __syncthreads();
    if (kk + 1 < H_DIM / 32) load_tiles((kk + 1) * 32);   // overlap with WMMAs

    FragBF fa[2];
#pragma unroll
    for (int im = 0; im < 2; ++im) {
      const int aoff = (mw * 32 + im * 16 + lrow) * LP + hi * 16;
      fa[im].q[0] = *(const uint4*)&sA[buf][aoff];
      fa[im].q[1] = *(const uint4*)&sA[buf][aoff + 8];
    }
#pragma unroll
    for (int f = 0; f < 4; ++f) {
      FragBF fg, fu;
      const int boff = (nw * 64 + f * 16 + lrow) * LP + hi * 16;
      fg.q[0] = *(const uint4*)&sBg[buf][boff];
      fg.q[1] = *(const uint4*)&sBg[buf][boff + 8];
      fu.q[0] = *(const uint4*)&sBu[buf][boff];
      fu.q[1] = *(const uint4*)&sBu[buf][boff + 8];
#pragma unroll
      for (int im = 0; im < 2; ++im) {
        accg[im][f] = wmma_bf16(fa[im], fg, accg[im][f]);
        accu[im][f] = wmma_bf16(fa[im], fu, accu[im][f]);
      }
    }
  }

  // epilogue: silu(g) * u -> bf16 inter
#pragma unroll
  for (int im = 0; im < 2; ++im) {
#pragma unroll
    for (int f = 0; f < 4; ++f) {
#pragma unroll
      for (int j = 0; j < 8; ++j) {
        const int m    = mw * 32 + im * 16 + 8 * hi + j;
        const int grow = mbase + m;
        if (grow < cnt) {
          const float g = accg[im][f][j];
          const float u = accu[im][f][j];
          const float s = g / (1.f + __expf(-g));
          inter[(size_t)(base + grow) * I_DIM + n0 + nw * 64 + f * 16 + lrow] = f2bf(s * u);
        }
      }
    }
  }
}

// ---------------------------------------------------------------- down GEMM
// out[token,:] += w_row * (inter[row,:] @ Wd[e]); block tile 128x128, K-step 32.
// A tile (already bf16) is staged with CDNA5 async global->LDS DMA (ASYNCcnt).
__global__ __launch_bounds__(256)
void moe_down_kernel(const u16* __restrict__ inter,
                     const float* __restrict__ Wd_all,
                     const int* __restrict__ cnts,
                     const int* __restrict__ bases,
                     const int* __restrict__ rowTok,
                     const float* __restrict__ rowW,
                     float* __restrict__ out)
{
  const int e     = blockIdx.z;
  const int cnt   = cnts[e];
  const int mbase = blockIdx.y * 128;
  if (mbase >= cnt) return;
  const int n0   = blockIdx.x * 128;
  const int base = bases[e];
  const float* Wd = Wd_all + (size_t)e * I_DIM * H_DIM;

  __shared__ u16   sA[2][128 * LP];
  __shared__ u16   sB[2][128 * LP];
  __shared__ int   sTok[128];
  __shared__ float sW[128];

  const int tid  = threadIdx.x;
  const int lane = tid & 31;
  const int wave = tid >> 5;
  const int lrow = lane & 15;
  const int hi   = lane >> 4;
  const int mw   = wave >> 1;
  const int nw   = wave & 1;

  if (tid < 128) {
    const int grow = mbase + tid;
    sTok[tid] = (grow < cnt) ? rowTok[base + grow] : 0;
    sW[tid]   = (grow < cnt) ? rowW[base + grow] : 0.f;
  }
  __syncthreads();

  // A source row, clamped into the valid inter range: rows beyond this expert's
  // count copy harmless garbage whose outputs are masked in the epilogue
  // (A row m only contributes to D row m).
  const int  totalRows = bases[E_NUM];
  const int  arow   = tid >> 1;
  const int  ahalf  = tid & 1;
  int arowg = base + mbase + arow;
  if (arowg >= totalRows) arowg = totalRows - 1;
  const u16* aptr = inter + (size_t)arowg * I_DIM + ahalf * 16;

  const int bn  = tid & 127;
  const int bkq = (tid >> 7) * 4;
  const float* dptr = Wd + n0 + bn;

  float df[16];

  // async DMA of this thread's 32B slice of the A tile straight into LDS
  auto async_load_A = [&](int buf, int K0) {
    const u16* g    = aptr + K0;
    const u32  lds0 = lds_addr_of(&sA[buf][arow * LP + ahalf * 16]);
    asm volatile("global_load_async_to_lds_b128 %0, %1, off"
                 :: "v"(lds0), "v"(g) : "memory");
    asm volatile("global_load_async_to_lds_b128 %0, %1, off"
                 :: "v"(lds0 + 16u), "v"(g + 8) : "memory");
  };

  auto load_B = [&](int K0) {
#pragma unroll
    for (int r = 0; r < 4; ++r) {
#pragma unroll
      for (int q = 0; q < 4; ++q) {
        df[4*r+q] = dptr[(size_t)(K0 + bkq + 8 * r + q) * H_DIM];
      }
    }
  };

  auto store_B = [&](int buf) {
#pragma unroll
    for (int r = 0; r < 4; ++r) {
      const int ks = bkq + 8 * r;
      *(uint2*)&sB[buf][bn * LP + ks] =
          make_uint2(pack2bf(df[4*r],df[4*r+1]), pack2bf(df[4*r+2],df[4*r+3]));
    }
  };

  v8f acc[2][4] = {};

  load_B(0);
  async_load_A(0, 0);
  for (int kk = 0; kk < I_DIM / 32; ++kk) {
    const int buf = kk & 1;
    store_B(buf);
    asm volatile("s_wait_asynccnt 0x0" ::: "memory");   // A tile DMA for `buf` done
    __syncthreads();
    if (kk + 1 < I_DIM / 32) {
      async_load_A(buf ^ 1, (kk + 1) * 32);             // DMA next tile under WMMAs
      load_B((kk + 1) * 32);
    }

    FragBF fa[2];
#pragma unroll
    for (int im = 0; im < 2; ++im) {
      const int aoff = (mw * 32 + im * 16 + lrow) * LP + hi * 16;
      fa[im].q[0] = *(const uint4*)&sA[buf][aoff];
      fa[im].q[1] = *(const uint4*)&sA[buf][aoff + 8];
    }
#pragma unroll
    for (int f = 0; f < 4; ++f) {
      FragBF fb;
      const int boff = (nw * 64 + f * 16 + lrow) * LP + hi * 16;
      fb.q[0] = *(const uint4*)&sB[buf][boff];
      fb.q[1] = *(const uint4*)&sB[buf][boff + 8];
#pragma unroll
      for (int im = 0; im < 2; ++im) {
        acc[im][f] = wmma_bf16(fa[im], fb, acc[im][f]);
      }
    }
  }

#pragma unroll
  for (int im = 0; im < 2; ++im) {
#pragma unroll
    for (int f = 0; f < 4; ++f) {
#pragma unroll
      for (int j = 0; j < 8; ++j) {
        const int m = mw * 32 + im * 16 + 8 * hi + j;
        if (mbase + m < cnt) {
          atomicAdd(&out[(size_t)sTok[m] * H_DIM + n0 + nw * 64 + f * 16 + lrow],
                    sW[m] * acc[im][f][j]);
        }
      }
    }
  }
}

// ---------------------------------------------------------------- launcher
extern "C" void kernel_launch(void* const* d_in, const int* in_sizes, int n_in,
                              void* d_out, int out_size, void* d_ws, size_t ws_size,
                              hipStream_t stream) {
  (void)in_sizes; (void)n_in; (void)out_size; (void)ws_size;
  const float* x  = (const float*)d_in[0];   // [T, H]
  const float* gw = (const float*)d_in[1];   // [H, E]
  const float* wg = (const float*)d_in[2];   // [E, H, I]
  const float* wu = (const float*)d_in[3];   // [E, H, I]
  const float* wd = (const float*)d_in[4];   // [E, I, H]

  float* out    = (float*)d_out;                       // [T, H]
  float* logits = out + (size_t)T_NUM * H_DIM;         // [T, E]

  char* ws = (char*)d_ws;
  int*   cnts    = (int*)  (ws + WS_COUNTS);
  int*   bases   = (int*)  (ws + WS_BASES);
  int*   tokSlot = (int*)  (ws + WS_TOKSLOT);
  int*   tokExp  = (int*)  (ws + WS_TOKEXP);
  float* tokW    = (float*)(ws + WS_TOKW);
  int*   rowTok  = (int*)  (ws + WS_ROWTOK);
  float* rowW    = (float*)(ws + WS_ROWW);
  u16*   inter   = (u16*)  (ws + WS_INTER);

  init_counts_kernel<<<1, 32, 0, stream>>>(cnts);
  zero_out_kernel<<<(T_NUM * H_DIM / 4) / 256, 256, 0, stream>>>((float4*)out);
  router_kernel<<<T_NUM / 8, 256, 0, stream>>>(x, gw, logits, cnts, tokSlot, tokExp, tokW);
  prefix_kernel<<<1, 1, 0, stream>>>(cnts, bases);
  scatter_kernel<<<T_NUM / 256, 256, 0, stream>>>(bases, tokSlot, tokExp, tokW, rowTok, rowW);

  moe_gateup_kernel<<<dim3(I_DIM / 128, T_NUM / 128, E_NUM), 256, 0, stream>>>(
      x, wg, wu, cnts, bases, rowTok, inter);
  moe_down_kernel<<<dim3(H_DIM / 128, T_NUM / 128, E_NUM), 256, 0, stream>>>(
      inter, wd, cnts, bases, rowTok, rowW, out);
}
